// CartesianMoEGenreClassifier_49331994362127
// MI455X (gfx1250) — matrix operand order repaired and checked by reading, not compiled, not run on hardware
//
#include <hip/hip_runtime.h>

typedef _Float16 h16;
typedef __attribute__((ext_vector_type(2)))  _Float16 v2h;
typedef __attribute__((ext_vector_type(16))) _Float16 v16h;
typedef __attribute__((ext_vector_type(8)))  float    v8f;

static constexpr int kD  = 768;
static constexpr int kH  = 12;
static constexpr int kL  = 6;
static constexpr int kFF = 3072;
static constexpr int kNA = 2;
static constexpr int kNB = 2;
static constexpr int kS  = 512;
static constexpr int kC  = 10;
static constexpr int kBS = 16;
static constexpr int kNE = 4;
static constexpr int kT  = kBS * kS;   // 8192 tokens
static constexpr int kDH = 64;         // head dim

// 1/sqrt(dh) * log2(e): fold into Q so softmax runs in the exp2 domain
#define KQSCALE 0.18033688011f

// ---------------------------------------------------------------- helpers

__device__ inline v8f wmma_f16(v16h a, v16h b, v8f c) {
  return __builtin_amdgcn_wmma_f32_16x16x32_f16(false, a, false, b, (short)0, c,
                                                false, false);
}

// DPP16 lane permute (VALU, no LDS). CTRL: 0xB1=xor1, 0x4E=xor2,
// 0x141=row_half_mirror (xor7), 0x140=row_mirror (xor15)
template <int CTRL>
__device__ inline float dppf(float x) {
  return __int_as_float(__builtin_amdgcn_update_dpp(
      0, __float_as_int(x), CTRL, 0xf, 0xf, true));
}

// 16-lane butterfly max via fused v_max_num_f32_dpp (1 VALU op per step,
// mask basis {1,2,7,15} spans the 16-lane group)
__device__ inline float redmax16(float v) {
  asm("v_max_num_f32_dpp %0, %1, %1 quad_perm:[1,0,3,2] row_mask:0xf bank_mask:0xf bound_ctrl:1"
      : "=v"(v) : "v"(v));
  asm("v_max_num_f32_dpp %0, %1, %1 quad_perm:[2,3,0,1] row_mask:0xf bank_mask:0xf bound_ctrl:1"
      : "=v"(v) : "v"(v));
  asm("v_max_num_f32_dpp %0, %1, %1 row_half_mirror row_mask:0xf bank_mask:0xf bound_ctrl:1"
      : "=v"(v) : "v"(v));
  asm("v_max_num_f32_dpp %0, %1, %1 row_mirror row_mask:0xf bank_mask:0xf bound_ctrl:1"
      : "=v"(v) : "v"(v));
  return v;
}
// 16-lane butterfly sum (compiler fuses these into v_add_f32_dpp)
__device__ inline float redsum16(float v) {
  v += dppf<0xB1>(v);
  v += dppf<0x4E>(v);
  v += dppf<0x141>(v);
  v += dppf<0x140>(v);
  return v;
}

// block-wide (256-thread) deterministic sum; sm must hold >= 8 floats
__device__ inline float blockSum256(float v, float* sm) {
  #pragma unroll
  for (int off = 1; off < 32; off <<= 1) v += __shfl_xor(v, off, 32);
  __syncthreads();
  if ((threadIdx.x & 31) == 0) sm[threadIdx.x >> 5] = v;
  __syncthreads();
  float r = 0.f;
  #pragma unroll
  for (int i = 0; i < 8; ++i) r += sm[i];
  return r;
}

// ---------------------------------------------------------------- f32 -> f16
__global__ void f2h_kernel(const float* __restrict__ a, h16* __restrict__ b, int n) {
  int i = blockIdx.x * 256 + threadIdx.x;
  if (i < n) b[i] = (h16)a[i];
}

// ---------------------------------------------------------------- embedding
__global__ void embed_kernel(const int* __restrict__ ids,
                             const float* __restrict__ tok,
                             const float* __restrict__ pos,
                             float* __restrict__ x, h16* __restrict__ xh) {
  int t = blockIdx.x;
  int id = ids[t];
  int s  = t & (kS - 1);
  size_t base = (size_t)t * kD;
  #pragma unroll
  for (int i = 0; i < 3; ++i) {
    int d = threadIdx.x + 256 * i;
    float v = tok[(size_t)id * kD + d] + pos[(size_t)s * kD + d];
    x[base + d]  = v;
    xh[base + d] = (h16)v;
  }
}

// ---------------------------------------------------------------- layernorm
__global__ void ln_kernel(float* __restrict__ x, const float* __restrict__ f,
                          const float* __restrict__ g, const float* __restrict__ bb,
                          h16* __restrict__ xh) {
  __shared__ float sm[8];
  int t = blockIdx.x;
  size_t base = (size_t)t * kD;
  int d0 = threadIdx.x, d1 = threadIdx.x + 256, d2 = threadIdx.x + 512;
  float v0 = x[base + d0] + f[base + d0];
  float v1 = x[base + d1] + f[base + d1];
  float v2 = x[base + d2] + f[base + d2];
  float mean = blockSum256(v0 + v1 + v2, sm) * (1.f / kD);
  float q0 = v0 - mean, q1 = v1 - mean, q2 = v2 - mean;
  float var = blockSum256(q0 * q0 + q1 * q1 + q2 * q2, sm) * (1.f / kD);
  float rstd = rsqrtf(var + 1e-5f);
  float y0 = q0 * rstd * g[d0] + bb[d0];
  float y1 = q1 * rstd * g[d1] + bb[d1];
  float y2 = q2 * rstd * g[d2] + bb[d2];
  x[base + d0] = y0; x[base + d1] = y1; x[base + d2] = y2;
  xh[base + d0] = (h16)y0; xh[base + d1] = (h16)y1; xh[base + d2] = (h16)y2;
}

// ---------------------------------------------------------------- WMMA GEMM
// out[m,n] = sum_k A[m,k] * W[n,k] + bias[n]
// block = 256 (8 waves as 2Mx4N), wave tile 16(M)x32(N) -> block tile 32x128
// grid.x = N/128, grid.y = Mmax/32
template <int GATHER>
__global__ void gemm_kernel(const h16* __restrict__ A, int lda,
                            const h16* __restrict__ W, int ldw,
                            const float* __restrict__ bias,
                            float* __restrict__ outF, h16* __restrict__ outH, int ldc,
                            int M, int K,
                            const int* __restrict__ list0,
                            const int* __restrict__ cntp,
                            const int* __restrict__ offp, int eid) {
  int lane = threadIdx.x & 31;
  int w = threadIdx.x >> 5;
  int mt = w >> 2, nt = w & 3;
  int cnt = M;
  const int* list = list0;
  if (GATHER) {
    cnt = cntp[eid];
    list = list0 + offp[eid];
    if ((int)(blockIdx.y * 32) >= cnt) return;
  }
  int m0 = blockIdx.y * 32 + mt * 16;
  int n0 = blockIdx.x * 128 + nt * 32;
  int r = lane & 15;
  int kh = (lane & 16) ? 16 : 0;
  int arow = m0 + r;
  if (GATHER) arow = list[arow < cnt ? arow : 0];
  const h16* Ab  = A + (size_t)arow * lda + ((lane & 16) ? 16 : 0);
  const h16* Wb0 = W + (size_t)(n0 + r) * ldw + kh;
  const h16* Wb1 = W + (size_t)(n0 + 16 + r) * ldw + kh;
  v8f c0 = {}, c1 = {};
  for (int k = 0; k < K; k += 32) {
    v16h a, b0, b1;
    #pragma unroll
    for (int j = 0; j < 8; ++j) { a[j] = Ab[k + j]; a[j + 8] = Ab[k + 8 + j]; }
    #pragma unroll
    for (int j = 0; j < 16; ++j) { b0[j] = Wb0[k + j]; b1[j] = Wb1[k + j]; }
    c0 = wmma_f16(a, b0, c0);
    c1 = wmma_f16(a, b1, c1);
  }
  float bn0 = bias ? bias[n0 + r] : 0.f;
  float bn1 = bias ? bias[n0 + 16 + r] : 0.f;
  int rb = (lane & 16) ? 8 : 0;
  #pragma unroll
  for (int q = 0; q < 8; ++q) {
    int m = m0 + rb + q;
    if (GATHER && m >= cnt) continue;
    int mrow = GATHER ? list[m] : m;
    size_t idx = (size_t)mrow * ldc + n0 + r;
    float v0 = c0[q] + bn0, v1 = c1[q] + bn1;
    if (outF) { outF[idx] = v0; outF[idx + 16] = v1; }
    if (outH) { outH[idx] = (h16)v0; outH[idx + 16] = (h16)v1; }
  }
}

// ---------------------------------------------------------------- dual GEMM + GLU
// g[m,n] = (A@W1[n].T + B1[n]) * silu(A@W1[FF+n].T + B1[FF+n]); grid.x = FF/128
template <int GATHER>
__global__ void glu_kernel(const h16* __restrict__ A, int lda,
                           const h16* __restrict__ W1, int ldw,
                           const float* __restrict__ B1,
                           h16* __restrict__ G, int ldg,
                           int M, int K,
                           const int* __restrict__ list0,
                           const int* __restrict__ cntp,
                           const int* __restrict__ offp, int eid) {
  int lane = threadIdx.x & 31;
  int w = threadIdx.x >> 5;
  int mt = w >> 2, nt = w & 3;
  int cnt = M;
  const int* list = list0;
  if (GATHER) {
    cnt = cntp[eid];
    list = list0 + offp[eid];
    if ((int)(blockIdx.y * 32) >= cnt) return;
  }
  int m0 = blockIdx.y * 32 + mt * 16;
  int n0 = blockIdx.x * 128 + nt * 32;
  int r = lane & 15;
  int kh = (lane & 16) ? 16 : 0;
  int arow = m0 + r;
  if (GATHER) arow = list[arow < cnt ? arow : 0];
  const h16* Ab  = A  + (size_t)arow * lda + ((lane & 16) ? 16 : 0);
  const h16* Wa0 = W1 + (size_t)(n0 + r) * ldw + kh;
  const h16* Wa1 = W1 + (size_t)(n0 + 16 + r) * ldw + kh;
  const h16* Wg0 = W1 + (size_t)(kFF + n0 + r) * ldw + kh;
  const h16* Wg1 = W1 + (size_t)(kFF + n0 + 16 + r) * ldw + kh;
  v8f ca0 = {}, ca1 = {}, cb0 = {}, cb1 = {};
  for (int k = 0; k < K; k += 32) {
    v16h a, b;
    #pragma unroll
    for (int j = 0; j < 8; ++j) { a[j] = Ab[k + j]; a[j + 8] = Ab[k + 8 + j]; }
    #pragma unroll
    for (int j = 0; j < 16; ++j) b[j] = Wa0[k + j];
    ca0 = wmma_f16(a, b, ca0);
    #pragma unroll
    for (int j = 0; j < 16; ++j) b[j] = Wa1[k + j];
    ca1 = wmma_f16(a, b, ca1);
    #pragma unroll
    for (int j = 0; j < 16; ++j) b[j] = Wg0[k + j];
    cb0 = wmma_f16(a, b, cb0);
    #pragma unroll
    for (int j = 0; j < 16; ++j) b[j] = Wg1[k + j];
    cb1 = wmma_f16(a, b, cb1);
  }
  float ba0 = B1[n0 + r],        ba1 = B1[n0 + 16 + r];
  float bg0 = B1[kFF + n0 + r],  bg1 = B1[kFF + n0 + 16 + r];
  int rb = (lane & 16) ? 8 : 0;
  #pragma unroll
  for (int q = 0; q < 8; ++q) {
    int m = m0 + rb + q;
    if (GATHER && m >= cnt) continue;
    int mrow = GATHER ? list[m] : m;
    float a0 = ca0[q] + ba0, g0 = cb0[q] + bg0;
    float a1 = ca1[q] + ba1, g1 = cb1[q] + bg1;
    float v0 = a0 * (g0 / (1.f + __expf(-g0)));
    float v1 = a1 * (g1 / (1.f + __expf(-g1)));
    size_t idx = (size_t)mrow * ldg + n0 + r;
    G[idx]      = (h16)v0;
    G[idx + 16] = (h16)v1;
  }
}

// ---------------------------------------------------------------- flash attention
// grid = (BS*H, S/128), block = 256 (8 waves, 16 query rows each)
// softmax runs in the exp2 domain: 1/sqrt(dh)*log2(e) is folded into Q.
// P@V is invariant under key permutation within a chunk, so the P tile is
// stored interleaved (slot 2r <- key r, slot 2r+1 <- key 16+r) to allow
// packed b32 LDS stores; V rows are read through the inverse slot->key map.
__global__ void attn_kernel(const h16* __restrict__ qkv, h16* __restrict__ O) {
  __shared__ h16 pb[8][16 * 32];
  int lane = threadIdx.x & 31;
  int w = threadIdx.x >> 5;
  int b = blockIdx.x / kH;
  int h = blockIdx.x % kH;
  int r = lane & 15;
  int hi = (lane >> 4) & 1;
  int qrow0 = b * kS + blockIdx.y * 128 + w * 16;
  const int ldq = 3 * kD;  // 2304
  const h16* Qp = qkv + (size_t)(qrow0 + r) * ldq + h * kDH + (hi ? 16 : 0);
  const h16* Kb = qkv + (size_t)(b * kS) * ldq + kD + h * kDH;
  const h16* Vb = qkv + (size_t)(b * kS) * ldq + 2 * kD + h * kDH;

  v16h qa0, qa1;
  #pragma unroll
  for (int j = 0; j < 8; ++j) {
    qa0[j]     = (h16)((float)Qp[j] * KQSCALE);
    qa0[j + 8] = (h16)((float)Qp[8 + j] * KQSCALE);
    qa1[j]     = (h16)((float)Qp[32 + j] * KQSCALE);
    qa1[j + 8] = (h16)((float)Qp[40 + j] * KQSCALE);
  }
  v8f oc[4] = {};
  float mrun[8], lrun[8];
  #pragma unroll
  for (int q = 0; q < 8; ++q) { mrun[q] = -1e30f; lrun[q] = 0.f; }
  h16* myp = &pb[w][0];

  for (int kc = 0; kc < kS; kc += 32) {
    const h16* K0 = Kb + (size_t)(kc + r) * ldq + (hi ? 16 : 0);
    const h16* K1 = Kb + (size_t)(kc + 16 + r) * ldq + (hi ? 16 : 0);
    v16h b00, b01, b10, b11;
    #pragma unroll
    for (int j = 0; j < 16; ++j) {
      b00[j] = K0[j]; b01[j] = K0[32 + j];
      b10[j] = K1[j]; b11[j] = K1[32 + j];
    }
    v8f s0 = {}, s1 = {};
    s0 = wmma_f16(qa0, b00, s0); s0 = wmma_f16(qa1, b01, s0);
    s1 = wmma_f16(qa0, b10, s1); s1 = wmma_f16(qa1, b11, s1);

    float p0[8], p1[8], scl[8];
    #pragma unroll
    for (int q = 0; q < 8; ++q) {
      float a0 = s0[q];          // already in log2 domain
      float a1 = s1[q];
      float tm = redmax16(fmaxf(a0, a1));   // fused v_max_num_f32_dpp butterfly
      float mnew = fmaxf(mrun[q], tm);
      scl[q] = __builtin_amdgcn_exp2f(mrun[q] - mnew);
      float e0 = __builtin_amdgcn_exp2f(a0 - mnew);
      float e1 = __builtin_amdgcn_exp2f(a1 - mnew);
      float rs = redsum16(e0 + e1);         // fused v_add_f32_dpp butterfly
      lrun[q] = lrun[q] * scl[q] + rs;
      mrun[q] = mnew;
      p0[q] = e0; p1[q] = e1;
    }
    #pragma unroll
    for (int nt = 0; nt < 4; ++nt)
      #pragma unroll
      for (int q = 0; q < 8; ++q) oc[nt][q] *= scl[q];

    // store P tile with interleaved key slots: one packed b32 per row
    #pragma unroll
    for (int q = 0; q < 8; ++q) {
      int row = hi * 8 + q;
      v2h pk;
      pk[0] = (h16)p0[q];
      pk[1] = (h16)p1[q];
      *(v2h*)&myp[row * 32 + 2 * r] = pk;
    }
    // reload as A-fragment (compiler inserts s_wait_dscnt for the dependency)
    v16h pa;
    #pragma unroll
    for (int j = 0; j < 8; ++j) {
      pa[j]     = myp[r * 32 + (hi ? 16 : 0) + j];
      pa[j + 8] = myp[r * 32 + 8 + (hi ? 16 : 0) + j];
    }
    #pragma unroll
    for (int nt = 0; nt < 4; ++nt) {
      v16h vb;
      #pragma unroll
      for (int j = 0; j < 16; ++j) {
        int slot = (hi ? 16 : 0) + j;
        int key  = ((slot & 1) ? 16 : 0) + (slot >> 1);  // inverse interleave
        vb[j] = Vb[(size_t)(kc + key) * ldq + nt * 16 + r];
      }
      oc[nt] = wmma_f16(pa, vb, oc[nt]);
    }
  }
  float linv[8];
  #pragma unroll
  for (int q = 0; q < 8; ++q) linv[q] = 1.f / lrun[q];
  #pragma unroll
  for (int nt = 0; nt < 4; ++nt)
    #pragma unroll
    for (int q = 0; q < 8; ++q) {
      int row = qrow0 + hi * 8 + q;
      O[(size_t)row * kD + h * kDH + nt * 16 + r] = (h16)(oc[nt][q] * linv[q]);
    }
}

// ---------------------------------------------------------------- MoE routing
__global__ void route_kernel(const float* __restrict__ x,
                             const float* __restrict__ gAw, const float* __restrict__ gAb,
                             const float* __restrict__ gBw, const float* __restrict__ gBb,
                             float* __restrict__ pA, float* __restrict__ pB,
                             int* __restrict__ eidx, int* __restrict__ cnt) {
  int lane = threadIdx.x & 31;
  int t = blockIdx.x * 8 + (threadIdx.x >> 5);
  const float* xr = x + (size_t)t * kD;
  float a0 = 0, a1 = 0, b0 = 0, b1 = 0;
  for (int d = lane; d < kD; d += 32) {
    float xv = xr[d];
    a0 += xv * gAw[d]; a1 += xv * gAw[kD + d];
    b0 += xv * gBw[d]; b1 += xv * gBw[kD + d];
  }
  #pragma unroll
  for (int off = 1; off < 32; off <<= 1) {
    a0 += __shfl_xor(a0, off, 32); a1 += __shfl_xor(a1, off, 32);
    b0 += __shfl_xor(b0, off, 32); b1 += __shfl_xor(b1, off, 32);
  }
  if (lane == 0) {
    a0 += gAb[0]; a1 += gAb[1]; b0 += gBb[0]; b1 += gBb[1];
    float m = fmaxf(a0, a1);
    float e0 = expf(a0 - m), e1 = expf(a1 - m), s = e0 + e1;
    float qa0 = e0 / s, qa1 = e1 / s;
    m = fmaxf(b0, b1);
    e0 = expf(b0 - m); e1 = expf(b1 - m); s = e0 + e1;
    float qb0 = e0 / s, qb1 = e1 / s;
    pA[2 * t] = qa0; pA[2 * t + 1] = qa1;
    pB[2 * t] = qb0; pB[2 * t + 1] = qb1;
    int iA = (qa0 >= qa1) ? 0 : 1;
    int iB = (qb0 >= qb1) ? 0 : 1;
    int e = iA * kNB + iB;
    eidx[t] = e;
    atomicAdd(&cnt[e], 1);
  }
}

__global__ void offs_kernel(const int* __restrict__ cnt, int* __restrict__ off) {
  if (threadIdx.x == 0) {
    int a = 0;
    for (int i = 0; i < kNE; ++i) { off[i] = a; a += cnt[i]; }
  }
}

__global__ void fill_kernel(const int* __restrict__ eidx, const int* __restrict__ off,
                            int* __restrict__ fill, int* __restrict__ list) {
  int t = blockIdx.x * 256 + threadIdx.x;
  if (t >= kT) return;
  int e = eidx[t];
  int p = atomicAdd(&fill[e], 1);
  list[off[e] + p] = t;
}

__global__ void aux_kernel(const float* __restrict__ pA, const float* __restrict__ pB,
                           float* __restrict__ auxAcc) {
  __shared__ float sm[8];
  float sa0 = 0, sa1 = 0, ca0 = 0, ca1 = 0, sb0 = 0, sb1 = 0, cb0 = 0, cb1 = 0;
  for (int t = threadIdx.x; t < kT; t += 256) {
    float a0 = pA[2 * t], a1 = pA[2 * t + 1];
    sa0 += a0; sa1 += a1;
    if (a0 >= a1) ca0 += 1.f; else ca1 += 1.f;
    float b0 = pB[2 * t], b1 = pB[2 * t + 1];
    sb0 += b0; sb1 += b1;
    if (b0 >= b1) cb0 += 1.f; else cb1 += 1.f;
  }
  sa0 = blockSum256(sa0, sm); sa1 = blockSum256(sa1, sm);
  ca0 = blockSum256(ca0, sm); ca1 = blockSum256(ca1, sm);
  sb0 = blockSum256(sb0, sm); sb1 = blockSum256(sb1, sm);
  cb0 = blockSum256(cb0, sm); cb1 = blockSum256(cb1, sm);
  if (threadIdx.x == 0) {
    const float invT = 1.f / kT;
    float auxA = kNA * ((sa0 * invT) * (ca0 * invT) + (sa1 * invT) * (ca1 * invT));
    float auxB = kNB * ((sb0 * invT) * (cb0 * invT) + (sb1 * invT) * (cb1 * invT));
    auxAcc[0] += auxA + auxB;
  }
}

// ---------------------------------------------------------------- head
__global__ void pool_kernel(const float* __restrict__ x, float* __restrict__ rep) {
  int b = blockIdx.x;
  #pragma unroll
  for (int i = 0; i < 3; ++i) {
    int d = threadIdx.x + 256 * i;
    float s = 0.f;
    for (int ss = 0; ss < kS; ++ss) s += x[(size_t)(b * kS + ss) * kD + d];
    rep[(size_t)b * kD + d] = s * (1.f / kS);
  }
}

__global__ void cls1_kernel(const float* __restrict__ rep, const float* __restrict__ W,
                            const float* __restrict__ bias, float* __restrict__ hcls) {
  int b = blockIdx.x;
  #pragma unroll
  for (int i = 0; i < 3; ++i) {
    int n = threadIdx.x + 256 * i;
    float s = bias[n];
    for (int d = 0; d < kD; ++d) s += rep[(size_t)b * kD + d] * W[(size_t)n * kD + d];
    hcls[(size_t)b * kD + n] = fmaxf(s, 0.f);
  }
}

__global__ void cls2_kernel(const float* __restrict__ hcls, const float* __restrict__ W,
                            const float* __restrict__ bias, float* __restrict__ out) {
  int i = threadIdx.x;
  if (i >= kBS * kC) return;
  int b = i / kC, c = i % kC;
  float s = bias[c];
  for (int d = 0; d < kD; ++d) s += hcls[(size_t)b * kD + d] * W[(size_t)c * kD + d];
  out[i] = s;
}

__global__ void finish_kernel(const float* __restrict__ auxAcc, float* __restrict__ out) {
  if (threadIdx.x == 0) out[kBS * kC] = auxAcc[0];
}

// ---------------------------------------------------------------- launch
extern "C" void kernel_launch(void* const* d_in, const int* in_sizes, int n_in,
                              void* d_out, int out_size, void* d_ws, size_t ws_size,
                              hipStream_t stream) {
  const int*   ids  = (const int*)  d_in[0];
  const float* tokE = (const float*)d_in[1];
  const float* posE = (const float*)d_in[2];
  const float* wqkv = (const float*)d_in[3];
  const float* bqkv = (const float*)d_in[4];
  const float* wo   = (const float*)d_in[5];
  const float* bo   = (const float*)d_in[6];
  const float* ln1g = (const float*)d_in[7];
  const float* ln1b = (const float*)d_in[8];
  const float* ln2g = (const float*)d_in[9];
  const float* ln2b = (const float*)d_in[10];
  const float* dW1  = (const float*)d_in[11];
  const float* dB1  = (const float*)d_in[12];
  const float* dW2  = (const float*)d_in[13];
  const float* dB2  = (const float*)d_in[14];
  const float* mW1  = (const float*)d_in[15];
  const float* mB1  = (const float*)d_in[16];
  const float* mW2  = (const float*)d_in[17];
  const float* mB2  = (const float*)d_in[18];
  const float* gAw  = (const float*)d_in[19];
  const float* gAb  = (const float*)d_in[20];
  const float* gBw  = (const float*)d_in[21];
  const float* gBb  = (const float*)d_in[22];
  const float* cW1  = (const float*)d_in[23];
  const float* cB1  = (const float*)d_in[24];
  const float* cW2  = (const float*)d_in[25];
  const float* cB2  = (const float*)d_in[26];
  float* out = (float*)d_out;

  char* p = (char*)d_ws;
  auto take = [&](size_t bytes) -> void* {
    void* r = (void*)p;
    p += (bytes + 255) & ~(size_t)255;
    return r;
  };
  float* xf    = (float*)take((size_t)kT * kD * 4);
  h16*   xh    = (h16*)  take((size_t)kT * kD * 2);
  h16*   qkvh  = (h16*)  take((size_t)kT * 3 * kD * 2);
  h16*   oh    = (h16*)  take((size_t)kT * kD * 2);
  float* buf1  = (float*)take((size_t)kT * kD * 4);
  h16*   gh    = (h16*)  take((size_t)kT * kFF * 2);
  h16*   wst1  = (h16*)  take((size_t)2 * kFF * kD * 2);
  h16*   wst2  = (h16*)  take((size_t)kFF * kD * 2);
  float* pAbuf = (float*)take((size_t)kT * 2 * 4);
  float* pBbuf = (float*)take((size_t)kT * 2 * 4);
  int*   eidx  = (int*)  take((size_t)kT * 4);
  int*   list  = (int*)  take((size_t)kT * 4);
  int*   cnt   = (int*)  take(kNE * 4);
  int*   fillA = (int*)  take(kNE * 4);
  int*   offp  = (int*)  take(kNE * 4);
  float* auxAc = (float*)take(4);
  float* rep   = (float*)take((size_t)kBS * kD * 4);
  float* hcls  = (float*)take((size_t)kBS * kD * 4);

  auto f2h = [&](const float* s, h16* d, int n) {
    f2h_kernel<<<dim3((n + 255) / 256), dim3(256), 0, stream>>>(s, d, n);
  };

  hipMemsetAsync(auxAc, 0, 4, stream);
  embed_kernel<<<kT, 256, 0, stream>>>(ids, tokE, posE, xf, xh);

  int di = 0, mi = 0;
  for (int l = 0; l < kL; ++l) {
    // ---- attention block
    f2h(wqkv + (size_t)l * 3 * kD * kD, wst1, 3 * kD * kD);
    gemm_kernel<0><<<dim3(3 * kD / 128, kT / 32), 256, 0, stream>>>(
        xh, kD, wst1, kD, bqkv + (size_t)l * 3 * kD,
        nullptr, qkvh, 3 * kD, kT, kD, nullptr, nullptr, nullptr, 0);
    attn_kernel<<<dim3(kBS * kH, kS / 128), 256, 0, stream>>>(qkvh, oh);
    f2h(wo + (size_t)l * kD * kD, wst1, kD * kD);
    gemm_kernel<0><<<dim3(kD / 128, kT / 32), 256, 0, stream>>>(
        oh, kD, wst1, kD, bo + (size_t)l * kD,
        buf1, nullptr, kD, kT, kD, nullptr, nullptr, nullptr, 0);
    ln_kernel<<<kT, 256, 0, stream>>>(xf, buf1, ln1g + (size_t)l * kD,
                                      ln1b + (size_t)l * kD, xh);

    // ---- FFN / MoE block
    bool moe = (l == 1) || (l == 3) || (l == 5);
    if (moe) {
      hipMemsetAsync(cnt, 0, kNE * 4, stream);
      hipMemsetAsync(fillA, 0, kNE * 4, stream);
      route_kernel<<<kT / 8, 256, 0, stream>>>(
          xf, gAw + (size_t)mi * kNA * kD, gAb + (size_t)mi * kNA,
          gBw + (size_t)mi * kNB * kD, gBb + (size_t)mi * kNB,
          pAbuf, pBbuf, eidx, cnt);
      offs_kernel<<<1, 32, 0, stream>>>(cnt, offp);
      fill_kernel<<<kT / 256, 256, 0, stream>>>(eidx, offp, fillA, list);
      aux_kernel<<<1, 256, 0, stream>>>(pAbuf, pBbuf, auxAc);
      for (int e = 0; e < kNE; ++e) {
        size_t xi = (size_t)(mi * kNE + e);
        f2h(mW1 + xi * 2 * kFF * kD, wst1, 2 * kFF * kD);
        glu_kernel<1><<<dim3(kFF / 128, kT / 32), 256, 0, stream>>>(
            xh, kD, wst1, kD, mB1 + xi * 2 * kFF,
            gh, kFF, 0, kD, list, cnt, offp, e);
        f2h(mW2 + xi * kD * kFF, wst2, kD * kFF);
        gemm_kernel<1><<<dim3(kD / 128, kT / 32), 256, 0, stream>>>(
            gh, kFF, wst2, kFF, mB2 + xi * kD,
            buf1, nullptr, kD, 0, kFF, list, cnt, offp, e);
      }
      ++mi;
    } else {
      f2h(dW1 + (size_t)di * 2 * kFF * kD, wst1, 2 * kFF * kD);
      glu_kernel<0><<<dim3(kFF / 128, kT / 32), 256, 0, stream>>>(
          xh, kD, wst1, kD, dB1 + (size_t)di * 2 * kFF,
          gh, kFF, kT, kD, nullptr, nullptr, nullptr, 0);
      f2h(dW2 + (size_t)di * kD * kFF, wst2, kD * kFF);
      gemm_kernel<0><<<dim3(kD / 128, kT / 32), 256, 0, stream>>>(
          gh, kFF, wst2, kFF, dB2 + (size_t)di * kD,
          buf1, nullptr, kD, kT, kFF, nullptr, nullptr, nullptr, 0);
      ++di;
    }
    ln_kernel<<<kT, 256, 0, stream>>>(xf, buf1, ln2g + (size_t)l * kD,
                                      ln2b + (size_t)l * kD, xh);
  }

  pool_kernel<<<kBS, 256, 0, stream>>>(xf, rep);
  cls1_kernel<<<kBS, 256, 0, stream>>>(rep, cW1, cB1, hcls);
  cls2_kernel<<<1, 256, 0, stream>>>(hcls, cW2, cB2, out);
  finish_kernel<<<1, 32, 0, stream>>>(auxAc, out);
}
